// Prompt_block_New_64355789963983
// MI455X (gfx1250) — compile-verified, matrix operand order checked by reading, not run on hardware
//
#include <hip/hip_runtime.h>
#include <hip/hip_bf16.h>

#define INPL  768
#define HIDE_ 64
#define BATCH 8
#define NPIX  1024          // 32*32
#define DSL   4
#define COUT  1536
#define KTOT  20736         // 768*27
#define NCOLS 32768         // B*D*NPIX

typedef __attribute__((ext_vector_type(8)))  __bf16 v8bf;
typedef __attribute__((ext_vector_type(16))) __bf16 v16bf;
typedef __attribute__((ext_vector_type(8)))  float  v8f;

union BF16Frag { v16bf v; v8bf h[2]; };

__device__ __forceinline__ unsigned short f2bf(float f) {
  unsigned u = __float_as_uint(f);
  u += 0x7FFFu + ((u >> 16) & 1u);          // round-to-nearest-even
  return (unsigned short)(u >> 16);
}
__device__ __forceinline__ float bf2f(unsigned short h) {
  return __uint_as_float(((unsigned)h) << 16);
}

// ---------------------------------------------------------------------------
// Stage 1: linear projections.
//  mode 0: q/k from rgb -> fp32 [b,n,h]
//  mode 2: v from hsi   -> bf16 TRANSPOSED vT[(d*B+b), h, m]  (m = pixel)
// block 256 (one n per thread), grid (4, B, 1 or D)
// ---------------------------------------------------------------------------
__global__ __launch_bounds__(256)
void pb_proj(const float* __restrict__ x, const float* __restrict__ Wm,
             float* __restrict__ dstF, unsigned short* __restrict__ dstH,
             int mode) {
  __shared__ float sw[96 * 64];               // W^T chunk [c][h], 24 KB
  const int tid = threadIdx.x;
  const int b = blockIdx.y, d = blockIdx.z;
  const int n = blockIdx.x * 256 + tid;
  const float* src;
  long cstride;
  if (mode == 0) { src = x + ((long)b * 3840) * 1024 + n;            cstride = 1024; }
  else           { src = x + ((long)b * 3840 + 768 + d) * 1024 + n;  cstride = 4096; }
  float acc[64];
#pragma unroll
  for (int h = 0; h < 64; ++h) acc[h] = 0.f;
  for (int c0 = 0; c0 < INPL; c0 += 96) {
    __syncthreads();
    for (int t = tid; t < 96 * 64; t += 256) {
      int cc = t >> 6, h = t & 63;
      sw[t] = Wm[h * INPL + c0 + cc];
    }
    __syncthreads();
    for (int cc = 0; cc < 96; ++cc) {
      float xv = src[(long)(c0 + cc) * cstride];
      const float* wr = &sw[cc * 64];
#pragma unroll
      for (int h = 0; h < 64; ++h) acc[h] += xv * wr[h];
    }
  }
  if (mode == 0) {
    float* o = dstF + ((long)b * NPIX + n) * 64;
#pragma unroll
    for (int h = 0; h < 64; ++h) o[h] = acc[h];
  } else {
    const long slab = (long)(d * BATCH + b);
#pragma unroll
    for (int h = 0; h < 64; ++h)
      dstH[(slab * 64 + h) * NPIX + n] = f2bf(acc[h]);
  }
}

// ---------------------------------------------------------------------------
// Stage 2: attn[b,i,:] = softmax_j( (q[b,i]·k[b,j]) * 64^-0.5 ), stored bf16
// one block (256 thr) per (i,b); each thread owns 4 j's
// ---------------------------------------------------------------------------
__global__ __launch_bounds__(256)
void pb_attn(const float* __restrict__ q, const float* __restrict__ k,
             unsigned short* __restrict__ attnb) {
  __shared__ float sq[64];
  __shared__ float red[256];
  const int tid = threadIdx.x;
  const int i = blockIdx.x, b = blockIdx.y;
  const float* qr = q + ((long)b * NPIX + i) * 64;
  if (tid < 64) sq[tid] = qr[tid];
  __syncthreads();
  float logit[4];
  float mx = -1e30f;
#pragma unroll
  for (int jj = 0; jj < 4; ++jj) {
    int j = jj * 256 + tid;
    const float* kr = k + ((long)b * NPIX + j) * 64;
    float s = 0.f;
#pragma unroll
    for (int h = 0; h < 64; ++h) s += sq[h] * kr[h];
    logit[jj] = s * 0.125f;
    mx = fmaxf(mx, logit[jj]);
  }
  red[tid] = mx; __syncthreads();
  for (int s = 128; s > 0; s >>= 1) { if (tid < s) red[tid] = fmaxf(red[tid], red[tid + s]); __syncthreads(); }
  const float m = red[0]; __syncthreads();
  float e[4], sum = 0.f;
#pragma unroll
  for (int jj = 0; jj < 4; ++jj) { e[jj] = __expf(logit[jj] - m); sum += e[jj]; }
  red[tid] = sum; __syncthreads();
  for (int s = 128; s > 0; s >>= 1) { if (tid < s) red[tid] += red[tid + s]; __syncthreads(); }
  const float inv = 1.0f / red[0];
  unsigned short* ar = attnb + ((long)b * NPIX + i) * NPIX;
#pragma unroll
  for (int jj = 0; jj < 4; ++jj) ar[jj * 256 + tid] = f2bf(e[jj] * inv);
}

// ---------------------------------------------------------------------------
// Stage 3 (WMMA): av[d,b,i,h] = sum_m attn[b,i,m] * v[d,b,m,h]
//   A = attnb rows (K contiguous), B = vT rows (K contiguous per h column).
// block 256 = 8 waves; wave owns 16 i-rows x 64 h (4 wmma accs), K=1024.
// grid (NPIX/128, B, D)
// ---------------------------------------------------------------------------
__global__ __launch_bounds__(256)
void pb_av_wmma(const unsigned short* __restrict__ attnb,
                const unsigned short* __restrict__ vT,
                float* __restrict__ av) {
  const int lane = threadIdx.x & 31;
  const int wave = threadIdx.x >> 5;
  const int b = blockIdx.y, d = blockIdx.z;
  const int i0 = blockIdx.x * 128 + wave * 16;
  const int hi = lane >> 4, lo = lane & 15, sel = hi * 8;
  const long slab = (long)(d * BATCH + b);
  const unsigned short* arow = attnb + ((long)b * NPIX + (i0 + lo)) * NPIX;
  const unsigned short* vbase = vT + slab * 64 * NPIX;

  v8f acc[4] = {};
  for (int kk = 0; kk < 32; ++kk) {
    const int k0 = kk * 32;
    BF16Frag a;
    a.h[0] = *(const v8bf*)(arow + k0 + sel);
    a.h[1] = *(const v8bf*)(arow + k0 + 16 + sel);
#pragma unroll
    for (int j = 0; j < 4; ++j) {
      BF16Frag bfr;
      const unsigned short* vr = vbase + (long)(j * 16 + lo) * NPIX + k0;
      bfr.h[0] = *(const v8bf*)(vr + sel);
      bfr.h[1] = *(const v8bf*)(vr + 16 + sel);
      acc[j] = __builtin_amdgcn_wmma_f32_16x16x32_bf16(false, a.v, false, bfr.v, (short)0, acc[j], false, false);
    }
  }
  float* ob = av + slab * NPIX * 64;
#pragma unroll
  for (int j = 0; j < 4; ++j)
#pragma unroll
    for (int rr = 0; rr < 8; ++rr)
      ob[(long)(i0 + rr + hi * 8) * 64 + j * 16 + lo] = acc[j][rr];
}

// ---------------------------------------------------------------------------
// Stage 4: scramble + @W1^T, emit activations as NDHWC bf16:
//   featT[b,d,n,c] = sum_h av[d,b,(n&15)*64+h, n>>4] * W1[c,h]
// one block per (n,b,d)
// ---------------------------------------------------------------------------
__global__ __launch_bounds__(256)
void pb_scrw1(const float* __restrict__ av, const float* __restrict__ W1,
              unsigned short* __restrict__ featT) {
  __shared__ float ss[64];
  const int tid = threadIdx.x;
  const int n = blockIdx.x, b = blockIdx.y, d = blockIdx.z;
  const float* avb = av + ((long)(d * BATCH + b)) * NPIX * 64;
  if (tid < 64)
    ss[tid] = avb[(long)((n & 15) * 64 + tid) * 64 + (n >> 4)];
  __syncthreads();
  unsigned short* orow = featT + (((long)(b * DSL + d)) * NPIX + n) * INPL;
  for (int c = tid; c < INPL; c += 256) {
    const float* w = W1 + (long)c * 64;
    float s = 0.f;
#pragma unroll
    for (int h = 0; h < 64; ++h) s += ss[h] * w[h];
    orow[c] = f2bf(s);
  }
}

// ---------------------------------------------------------------------------
// Stage 5: repack conv weights to bf16 GEMM-A: Aw[co][r*768+ci]
// ---------------------------------------------------------------------------
__global__ __launch_bounds__(256)
void pb_repack(const float* __restrict__ conv_w, unsigned short* __restrict__ Aw) {
  long t = (long)blockIdx.x * 256 + threadIdx.x;
  if (t >= (long)COUT * KTOT) return;
  int co = (int)(t / KTOT);
  int kk = (int)(t % KTOT);
  int r = kk / INPL, ci = kk % INPL;
  Aw[t] = f2bf(conv_w[(long)co * KTOT + (long)ci * 27 + r]);
}

// ---------------------------------------------------------------------------
// Stage 6: Conv3d as implicit GEMM via v_wmma_f32_16x16x32_bf16.
// WG 256 thr = 8 waves (2 co x 4 col); wave tile 32co x 64col = 2x4 WMMA accs
// (8 wmma per 32-deep K step from 2 A + 4 B fragments).
// Halo handled by per-lane predicated B loads (zero fragments OOB).
// Gate nonlinearities fused into epilogue -> zf[col][co] bf16.
// grid (COUT/64, NCOLS/256)
// ---------------------------------------------------------------------------
__global__ __launch_bounds__(256)
void pb_conv3d_wmma(const unsigned short* __restrict__ Aw,
                    const unsigned short* __restrict__ featT,
                    const float* __restrict__ conv_b,
                    unsigned short* __restrict__ zf) {
  const int lane  = threadIdx.x & 31;
  const int wave  = threadIdx.x >> 5;
  const int waveM = wave >> 2;          // 0..1
  const int waveN = wave & 3;           // 0..3
  const int coBase  = blockIdx.x * 64 + waveM * 32;
  const int colBase = blockIdx.y * 256 + waveN * 64;

  const int hi  = lane >> 4;            // half-wave select
  const int lo  = lane & 15;
  const int sel = hi * 8;               // K offset per ISA 16-bit A/B layout

  const long rowA0 = (long)(coBase + lo) * KTOT;
  const long rowA1 = (long)(coBase + 16 + lo) * KTOT;

  int bx[4], dx[4], yx[4], xx[4];
#pragma unroll
  for (int j = 0; j < 4; ++j) {
    const int gcol = colBase + j * 16 + lo;
    bx[j] = gcol >> 12; dx[j] = (gcol >> 10) & 3;
    yx[j] = (gcol >> 5) & 31; xx[j] = gcol & 31;
  }

  v8f acc[2][4] = {};

  for (int r = 0; r < 27; ++r) {
    const int kd = r / 9;
    const int ky = (r / 3) % 3;
    const int kx = r % 3;
    bool ok[4];
    long base[4];
#pragma unroll
    for (int j = 0; j < 4; ++j) {
      const int dd = dx[j] + kd - 1, yy = yx[j] + ky - 1, xq = xx[j] + kx - 1;
      ok[j] = ((unsigned)dd < 4u) && ((unsigned)yy < 32u) && ((unsigned)xq < 32u);
      base[j] = ok[j] ? ((long)(((bx[j] * 4 + dd) << 10) + (yy << 5) + xq)) * INPL : 0;
    }
    const long aOff = (long)r * INPL;

    for (int cc = 0; cc < 24; ++cc) {
      const int ci0 = cc * 32;
      BF16Frag a0, a1;
      a0.h[0] = *(const v8bf*)(Aw + rowA0 + aOff + ci0 + sel);
      a0.h[1] = *(const v8bf*)(Aw + rowA0 + aOff + ci0 + 16 + sel);
      a1.h[0] = *(const v8bf*)(Aw + rowA1 + aOff + ci0 + sel);
      a1.h[1] = *(const v8bf*)(Aw + rowA1 + aOff + ci0 + 16 + sel);
      BF16Frag bb[4];
#pragma unroll
      for (int j = 0; j < 4; ++j) {
        bb[j].v = a0.v;                 // overwritten below; keep defined
        if (ok[j]) {
          bb[j].h[0] = *(const v8bf*)(featT + base[j] + ci0 + sel);
          bb[j].h[1] = *(const v8bf*)(featT + base[j] + ci0 + 16 + sel);
        } else {
          BF16Frag z = {};
          bb[j] = z;
        }
      }
#pragma unroll
      for (int j = 0; j < 4; ++j) {
        acc[0][j] = __builtin_amdgcn_wmma_f32_16x16x32_bf16(false, a0.v, false, bb[j].v, (short)0, acc[0][j], false, false);
        acc[1][j] = __builtin_amdgcn_wmma_f32_16x16x32_bf16(false, a1.v, false, bb[j].v, (short)0, acc[1][j], false, false);
      }
    }
  }

  // Epilogue: bias + tanh/sigmoid gates, store bf16 to zf[col][co]
#pragma unroll
  for (int i = 0; i < 2; ++i) {
#pragma unroll
    for (int j = 0; j < 4; ++j) {
      const int colT = colBase + j * 16 + lo;           // D layout: N = lane&15
      const long orow = (long)colT * COUT;
#pragma unroll
      for (int rr = 0; rr < 8; ++rr) {
        const int co = coBase + i * 16 + rr + hi * 8;   // D layout: M = rr + 8*(lane>>4)
        float g = acc[i][j][rr] + conv_b[co];
        float val = (co < INPL) ? tanhf(g) : (1.0f / (1.0f + __expf(-g)));
        zf[orow + co] = f2bf(val);
      }
    }
  }
}

// ---------------------------------------------------------------------------
// Stage 7: fQRNN scan over depth (h = f*h + (1-f)*z), or passthrough if i!=0
// ---------------------------------------------------------------------------
__global__ __launch_bounds__(256)
void pb_scan(const unsigned short* __restrict__ zf,
             const unsigned short* __restrict__ featT,
             const int* __restrict__ iptr, float* __restrict__ out) {
  const int c = blockIdx.x * 256 + threadIdx.x;
  const int n = blockIdx.y, b = blockIdx.z;
  float val;
  if (*iptr != 0) {
    val = bf2f(featT[(((long)(b * DSL + 3)) * NPIX + n) * INPL + c]);  // prompts[-1]
  } else {
    float h = 0.f;
    for (int d = 0; d < DSL; ++d) {
      const long row = (((long)(b * DSL + d)) * NPIX + n) * COUT;
      float z = bf2f(zf[row + c]);
      float f = bf2f(zf[row + INPL + c]);
      h = f * h + (1.f - f) * z;
    }
    val = h;
  }
  out[((long)b * INPL + c) * NPIX + n] = val;
}

// ---------------------------------------------------------------------------
extern "C" void kernel_launch(void* const* d_in, const int* in_sizes, int n_in,
                              void* d_out, int out_size, void* d_ws, size_t ws_size,
                              hipStream_t stream) {
  (void)in_sizes; (void)n_in; (void)out_size; (void)ws_size;
  const float* x      = (const float*)d_in[0];
  const float* Wq     = (const float*)d_in[1];
  const float* Wk     = (const float*)d_in[2];
  const float* Wv     = (const float*)d_in[3];
  const float* W1     = (const float*)d_in[4];
  const float* conv_w = (const float*)d_in[5];
  const float* conv_b = (const float*)d_in[6];
  const int*   ip     = (const int*)d_in[7];
  float* out = (float*)d_out;

  char* w = (char*)d_ws;
  size_t off = 0;
  auto alloc = [&](size_t bytes) -> void* {
    void* p = w + off;
    off += (bytes + 255) & ~(size_t)255;
    return p;
  };
  float* q    = (float*)alloc((size_t)BATCH * NPIX * HIDE_ * 4);
  float* k    = (float*)alloc((size_t)BATCH * NPIX * HIDE_ * 4);
  unsigned short* attnb = (unsigned short*)alloc((size_t)BATCH * NPIX * NPIX * 2);
  unsigned short* vT    = (unsigned short*)alloc((size_t)DSL * BATCH * HIDE_ * NPIX * 2);
  float* av   = (float*)alloc((size_t)DSL * BATCH * NPIX * HIDE_ * 4);
  unsigned short* featT = (unsigned short*)alloc((size_t)NCOLS * INPL * 2);
  unsigned short* Aw    = (unsigned short*)alloc((size_t)COUT * KTOT * 2);
  unsigned short* zf    = (unsigned short*)alloc((size_t)NCOLS * COUT * 2);

  pb_repack<<<dim3((unsigned)(((long)COUT * KTOT + 255) / 256)), 256, 0, stream>>>(conv_w, Aw);
  pb_proj<<<dim3(4, BATCH, 1), 256, 0, stream>>>(x, Wq, q, nullptr, 0);
  pb_proj<<<dim3(4, BATCH, 1), 256, 0, stream>>>(x, Wk, k, nullptr, 0);
  pb_proj<<<dim3(4, BATCH, DSL), 256, 0, stream>>>(x, Wv, nullptr, vT, 2);
  pb_attn<<<dim3(NPIX, BATCH), 256, 0, stream>>>(q, k, attnb);
  pb_av_wmma<<<dim3(NPIX / 128, BATCH, DSL), 256, 0, stream>>>(attnb, vT, av);
  pb_scrw1<<<dim3(NPIX, BATCH, DSL), 256, 0, stream>>>(av, W1, featT);
  pb_conv3d_wmma<<<dim3(COUT / 64, NCOLS / 256), 256, 0, stream>>>(Aw, featT, conv_b, zf);
  pb_scan<<<dim3(INPL / 256, NPIX, BATCH), 256, 0, stream>>>(zf, featT, ip, out);
}
